// AutoregressiveDecoder_69252052681264
// MI455X (gfx1250) — matrix operand order repaired and checked by reading, not compiled, not run on hardware
//
#include <hip/hip_runtime.h>

#define DIMC 256
#define NHEADS 8
#define DHEAD 32
#define NE 97
#define NTT 20
#define NS (NTT * NE)   /* 1940 */
#define S_PAD 1952      /* 122 * 16 */
#define FFN_DIM 1024
#define NLAYERS 6

typedef _Float16 v16h __attribute__((ext_vector_type(16)));
typedef float v8f __attribute__((ext_vector_type(8)));

union Frag { v16h v; unsigned u[8]; };
union Half8 { _Float16 h[8]; uint4 q; };

// ---------------------------------------------------------------------------
// Weight prep: W (f32, [L][K][N] row-major)  ->  Bp (f16, [L][N][K])
// so that per-lane WMMA B-fragment dwords (k,k+1 pairs at fixed n) are
// contiguous aligned 32-bit loads (compiler merges them into b128).
// ---------------------------------------------------------------------------
__global__ void prep_weight_kernel(const float* __restrict__ W,
                                   _Float16* __restrict__ Bp, int K, int N) {
    size_t base = (size_t)blockIdx.y * K * N;
    int i = blockIdx.x * blockDim.x + threadIdx.x;
    if (i < K * N) {
        int n = i / K, k = i - n * K;
        Bp[base + i] = (_Float16)W[base + (size_t)k * N + n];
    }
}

// ---------------------------------------------------------------------------
// Embedding add + f16 shadow copy; zero the pad rows (S..S_PAD).
// ---------------------------------------------------------------------------
__global__ __launch_bounds__(DIMC) void embed_kernel(
    const float* __restrict__ x, const float* __restrict__ sp,
    const float* __restrict__ te, float* __restrict__ x32,
    _Float16* __restrict__ x16) {
    int row = blockIdx.x, c = threadIdx.x;
    float v = 0.f;
    if (row < NS) {
        int t = row / NE, e = row - t * NE;
        v = x[(size_t)row * DIMC + c] + sp[(size_t)e * DIMC + c] +
            te[(size_t)t * DIMC + c];
    }
    x32[(size_t)row * DIMC + c] = v;
    x16[(size_t)row * DIMC + c] = (_Float16)v;
}

// ---------------------------------------------------------------------------
// WMMA GEMM:  out[M,N] = A16[M,K] * Bp16[N,K]^T + bias  (opt ReLU)
// OMODE: 0 = f32 row-major, 1 = f16 row-major, 2 = f16 transposed [N][S_PAD]
// Wave computes a 16x64 tile; 4 waves/block cover 16 rows x 256 cols.
// grid = (S_PAD/16, N/256), block = 128.
// ---------------------------------------------------------------------------
template <bool RELU, int OMODE>
__global__ __launch_bounds__(128) void gemm_kernel(
    const _Float16* __restrict__ A, const _Float16* __restrict__ B,
    const float* __restrict__ bias, float* __restrict__ outF,
    _Float16* __restrict__ outH, int K, int N) {
    const int lane = threadIdx.x & 31;
    const int wave = threadIdx.x >> 5;
    const int col = lane & 15;
    const int half = lane >> 4;
    const int mbase = blockIdx.x * 16;
    const int nbase = blockIdx.y * 256 + wave * 64;

    const _Float16* aptr = A + (size_t)(mbase + col) * K + half * 16;
    const _Float16* bptr = B + (size_t)(nbase + col) * K + half * 16;

    // one-shot pull of the weight columns toward the WGP (global_prefetch_b8);
    // keep it out of the K-loop so the loop still unrolls cleanly.
    __builtin_prefetch(bptr, 0, 1);
    __builtin_prefetch(bptr + (size_t)2 * 16 * K, 0, 1);

    v8f acc0 = {}, acc1 = {}, acc2 = {}, acc3 = {};
#pragma unroll 2
    for (int k0 = 0; k0 < K; k0 += 32) {
        Frag af;
#pragma unroll
        for (int j = 0; j < 4; ++j) {
            af.u[j]     = *(const unsigned*)(aptr + k0 + 2 * j);
            af.u[4 + j] = *(const unsigned*)(aptr + k0 + 8 + 2 * j);
        }
        Frag bf0, bf1, bf2, bf3;
#pragma unroll
        for (int j = 0; j < 8; ++j) {
            bf0.u[j] = *(const unsigned*)(bptr + (size_t)0 * 16 * K + k0 + 2 * j);
            bf1.u[j] = *(const unsigned*)(bptr + (size_t)1 * 16 * K + k0 + 2 * j);
            bf2.u[j] = *(const unsigned*)(bptr + (size_t)2 * 16 * K + k0 + 2 * j);
            bf3.u[j] = *(const unsigned*)(bptr + (size_t)3 * 16 * K + k0 + 2 * j);
        }
        acc0 = __builtin_amdgcn_wmma_f32_16x16x32_f16(false, af.v, false, bf0.v,
                                                      (short)0, acc0, false, false);
        acc1 = __builtin_amdgcn_wmma_f32_16x16x32_f16(false, af.v, false, bf1.v,
                                                      (short)0, acc1, false, false);
        acc2 = __builtin_amdgcn_wmma_f32_16x16x32_f16(false, af.v, false, bf2.v,
                                                      (short)0, acc2, false, false);
        acc3 = __builtin_amdgcn_wmma_f32_16x16x32_f16(false, af.v, false, bf3.v,
                                                      (short)0, acc3, false, false);
    }
    v8f accs[4] = {acc0, acc1, acc2, acc3};
#pragma unroll
    for (int nt = 0; nt < 4; ++nt) {
        int n = nbase + nt * 16 + col;
        float bv = bias[n];
        if (OMODE == 2) {
            // 8 rows (mbase+8*half .. +7) are contiguous in [N][S_PAD]:
            // pack into one aligned 16-byte store.
            Half8 o;
#pragma unroll
            for (int r = 0; r < 8; ++r) {
                float v = accs[nt][r] + bv;
                if (RELU) v = fmaxf(v, 0.f);
                o.h[r] = (_Float16)v;
            }
            *(uint4*)(outH + (size_t)n * S_PAD + mbase + 8 * half) = o.q;
        } else {
#pragma unroll
            for (int r = 0; r < 8; ++r) {
                int row = mbase + r + 8 * half;
                float v = accs[nt][r] + bv;
                if (RELU) v = fmaxf(v, 0.f);
                if (OMODE == 0)
                    outF[(size_t)row * N + n] = v;
                else
                    outH[(size_t)row * N + n] = (_Float16)v;
            }
        }
    }
}

// ---------------------------------------------------------------------------
// Flash attention: one wave per (16-query tile, head). Block-causal mask.
// q16,k16: [S_PAD][256] f16;  vT: [256][S_PAD] f16;  attn16 out: [S_PAD][256]
// grid = (S_PAD/16, HEADS), block = 32.
// ---------------------------------------------------------------------------
__global__ __launch_bounds__(32) void attn_kernel(
    const _Float16* __restrict__ q16, const _Float16* __restrict__ k16,
    const _Float16* __restrict__ vT, _Float16* __restrict__ attn16) {
    __shared__ _Float16 pLds[16 * 32];
    const int lane = threadIdx.x;
    const int col = lane & 15;
    const int half = lane >> 4;
    const int h = blockIdx.y;
    const int qbase = blockIdx.x * 16;

    // Q A-fragment (K = head dim 32)
    Frag qf;
    {
        const _Float16* qp =
            q16 + (size_t)(qbase + col) * DIMC + h * DHEAD + half * 16;
#pragma unroll
        for (int j = 0; j < 4; ++j) {
            qf.u[j]     = *(const unsigned*)(qp + 2 * j);
            qf.u[4 + j] = *(const unsigned*)(qp + 8 + 2 * j);
        }
    }
    int qlim[8];
#pragma unroll
    for (int r = 0; r < 8; ++r) {
        int row = qbase + r + 8 * half;
        int rc = row < NS ? row : NS - 1;
        qlim[r] = (rc / NE + 1) * NE;
    }
    int lastrow = (qbase + 15 < NS) ? qbase + 15 : NS - 1;
    int nblk = ((lastrow / NE + 1) * NE + 31) / 32;

    float m[8], l[8];
    v8f acc0 = {}, acc1 = {};
#pragma unroll
    for (int r = 0; r < 8; ++r) { m[r] = -3.0e38f; l[r] = 0.f; }
    const float scale = 0.17677669529663687f;  // 1/sqrt(32)

    for (int kb = 0; kb < nblk; ++kb) {
        int k0 = kb * 32;
        // K^T B-fragments: two tiles of 16 keys
        Frag kf0, kf1;
        {
            const _Float16* kp0 =
                k16 + (size_t)(k0 + col) * DIMC + h * DHEAD + half * 16;
            const _Float16* kp1 = kp0 + (size_t)16 * DIMC;
#pragma unroll
            for (int j = 0; j < 8; ++j) {
                kf0.u[j] = *(const unsigned*)(kp0 + 2 * j);
                kf1.u[j] = *(const unsigned*)(kp1 + 2 * j);
            }
        }
        v8f z = {};
        v8f s0 = __builtin_amdgcn_wmma_f32_16x16x32_f16(false, qf.v, false, kf0.v,
                                                        (short)0, z, false, false);
        v8f s1 = __builtin_amdgcn_wmma_f32_16x16x32_f16(false, qf.v, false, kf1.v,
                                                        (short)0, z, false, false);
        int key0 = k0 + col, key1 = k0 + 16 + col;
        float p0[8], p1[8], alpha[8];
#pragma unroll
        for (int r = 0; r < 8; ++r) {
            float a = s0[r] * scale; if (key0 >= qlim[r]) a = -1e9f;
            float b = s1[r] * scale; if (key1 >= qlim[r]) b = -1e9f;
            float rm = fmaxf(a, b);
#pragma unroll
            for (int sh = 8; sh >= 1; sh >>= 1) rm = fmaxf(rm, __shfl_xor(rm, sh, 32));
            float mn = fmaxf(m[r], rm);
            alpha[r] = __expf(m[r] - mn);
            float pa = __expf(a - mn), pb = __expf(b - mn);
            float rs = pa + pb;
#pragma unroll
            for (int sh = 8; sh >= 1; sh >>= 1) rs += __shfl_xor(rs, sh, 32);
            l[r] = l[r] * alpha[r] + rs;
            m[r] = mn;
            p0[r] = pa; p1[r] = pb;
        }
#pragma unroll
        for (int r = 0; r < 8; ++r) { acc0[r] *= alpha[r]; acc1[r] *= alpha[r]; }
        // Stage P (16x32) in LDS to reshape D-layout -> A-fragment layout
#pragma unroll
        for (int r = 0; r < 8; ++r) {
            int row = r + 8 * half;
            pLds[row * 32 + col] = (_Float16)p0[r];
            pLds[row * 32 + 16 + col] = (_Float16)p1[r];
        }
        __syncthreads();
        Frag pf;
        {
            const _Float16* pp = pLds + col * 32 + half * 16;
#pragma unroll
            for (int j = 0; j < 4; ++j) {
                pf.u[j]     = *(const unsigned*)(pp + 2 * j);
                pf.u[4 + j] = *(const unsigned*)(pp + 8 + 2 * j);
            }
        }
        // V B-fragments: d-tiles 0..15 and 16..31, K = 32 keys (contiguous in vT)
        Frag vf0, vf1;
        {
            const _Float16* vp0 =
                vT + (size_t)(h * DHEAD + col) * S_PAD + k0 + half * 16;
            const _Float16* vp1 = vp0 + (size_t)16 * S_PAD;
#pragma unroll
            for (int j = 0; j < 8; ++j) {
                vf0.u[j] = *(const unsigned*)(vp0 + 2 * j);
                vf1.u[j] = *(const unsigned*)(vp1 + 2 * j);
            }
        }
        acc0 = __builtin_amdgcn_wmma_f32_16x16x32_f16(false, pf.v, false, vf0.v,
                                                      (short)0, acc0, false, false);
        acc1 = __builtin_amdgcn_wmma_f32_16x16x32_f16(false, pf.v, false, vf1.v,
                                                      (short)0, acc1, false, false);
        __syncthreads();
    }
#pragma unroll
    for (int r = 0; r < 8; ++r) {
        int row = qbase + r + 8 * half;
        float inv = l[r] > 0.f ? 1.f / l[r] : 0.f;
        attn16[(size_t)row * DIMC + h * DHEAD + col] = (_Float16)(acc0[r] * inv);
        attn16[(size_t)row * DIMC + h * DHEAD + 16 + col] = (_Float16)(acc1[r] * inv);
    }
}

// ---------------------------------------------------------------------------
// Residual add + LayerNorm; writes f32 state and f16 shadow. grid=NS, block=256
// ---------------------------------------------------------------------------
__global__ __launch_bounds__(DIMC) void ln_residual_kernel(
    const float* xin, const float* __restrict__ y, const float* __restrict__ g,
    const float* __restrict__ b, float* xout, _Float16* __restrict__ x16out) {
    __shared__ float red[8];
    int row = blockIdx.x, c = threadIdx.x;
    float v = xin[(size_t)row * DIMC + c] + y[(size_t)row * DIMC + c];
    float s = v;
#pragma unroll
    for (int mm = 16; mm >= 1; mm >>= 1) s += __shfl_xor(s, mm, 32);
    if ((c & 31) == 0) red[c >> 5] = s;
    __syncthreads();
    float mean = 0.f;
#pragma unroll
    for (int i = 0; i < 8; ++i) mean += red[i];
    mean *= (1.f / DIMC);
    float d = v - mean;
    float sq = d * d;
#pragma unroll
    for (int mm = 16; mm >= 1; mm >>= 1) sq += __shfl_xor(sq, mm, 32);
    __syncthreads();
    if ((c & 31) == 0) red[c >> 5] = sq;
    __syncthreads();
    float var = 0.f;
#pragma unroll
    for (int i = 0; i < 8; ++i) var += red[i];
    var *= (1.f / DIMC);
    float o = d * rsqrtf(var + 1e-5f) * g[c] + b[c];
    xout[(size_t)row * DIMC + c] = o;
    x16out[(size_t)row * DIMC + c] = (_Float16)o;
}

// ---------------------------------------------------------------------------
static inline char* wsalloc(char*& w, size_t bytes) {
    char* p = w;
    w += (bytes + 255) & ~(size_t)255;
    return p;
}

extern "C" void kernel_launch(void* const* d_in, const int* in_sizes, int n_in,
                              void* d_out, int out_size, void* d_ws,
                              size_t ws_size, hipStream_t stream) {
    const float* x   = (const float*)d_in[0];
    const float* sp  = (const float*)d_in[1];
    const float* te  = (const float*)d_in[2];
    const float* Wq  = (const float*)d_in[3];
    const float* Wk  = (const float*)d_in[4];
    const float* Wv  = (const float*)d_in[5];
    const float* Wo  = (const float*)d_in[6];
    const float* bq  = (const float*)d_in[7];
    const float* bk  = (const float*)d_in[8];
    const float* bv  = (const float*)d_in[9];
    const float* bo  = (const float*)d_in[10];
    const float* W1  = (const float*)d_in[11];
    const float* b1  = (const float*)d_in[12];
    const float* W2  = (const float*)d_in[13];
    const float* b2  = (const float*)d_in[14];
    const float* ln1g = (const float*)d_in[15];
    const float* ln1b = (const float*)d_in[16];
    const float* ln2g = (const float*)d_in[17];
    const float* ln2b = (const float*)d_in[18];
    float* out = (float*)d_out;

    char* w = (char*)d_ws;
    float*    x32  = (float*)wsalloc(w, (size_t)S_PAD * DIMC * 4);
    float*    y32  = (float*)wsalloc(w, (size_t)S_PAD * DIMC * 4);
    _Float16* x16  = (_Float16*)wsalloc(w, (size_t)S_PAD * DIMC * 2);
    _Float16* q16  = (_Float16*)wsalloc(w, (size_t)S_PAD * DIMC * 2);
    _Float16* k16  = (_Float16*)wsalloc(w, (size_t)S_PAD * DIMC * 2);
    _Float16* vT16 = (_Float16*)wsalloc(w, (size_t)DIMC * S_PAD * 2);
    _Float16* at16 = (_Float16*)wsalloc(w, (size_t)S_PAD * DIMC * 2);
    _Float16* h16  = (_Float16*)wsalloc(w, (size_t)S_PAD * FFN_DIM * 2);
    _Float16* WqT  = (_Float16*)wsalloc(w, (size_t)NLAYERS * DIMC * DIMC * 2);
    _Float16* WkT  = (_Float16*)wsalloc(w, (size_t)NLAYERS * DIMC * DIMC * 2);
    _Float16* WvT  = (_Float16*)wsalloc(w, (size_t)NLAYERS * DIMC * DIMC * 2);
    _Float16* WoT  = (_Float16*)wsalloc(w, (size_t)NLAYERS * DIMC * DIMC * 2);
    _Float16* W1T  = (_Float16*)wsalloc(w, (size_t)NLAYERS * DIMC * FFN_DIM * 2);
    _Float16* W2T  = (_Float16*)wsalloc(w, (size_t)NLAYERS * FFN_DIM * DIMC * 2);

    // 1) weight conversion / transposition to f16 [N][K]
    {
        dim3 g1((DIMC * DIMC + 255) / 256, NLAYERS);
        prep_weight_kernel<<<g1, 256, 0, stream>>>(Wq, WqT, DIMC, DIMC);
        prep_weight_kernel<<<g1, 256, 0, stream>>>(Wk, WkT, DIMC, DIMC);
        prep_weight_kernel<<<g1, 256, 0, stream>>>(Wv, WvT, DIMC, DIMC);
        prep_weight_kernel<<<g1, 256, 0, stream>>>(Wo, WoT, DIMC, DIMC);
        dim3 g2((DIMC * FFN_DIM + 255) / 256, NLAYERS);
        prep_weight_kernel<<<g2, 256, 0, stream>>>(W1, W1T, DIMC, FFN_DIM);
        prep_weight_kernel<<<g2, 256, 0, stream>>>(W2, W2T, FFN_DIM, DIMC);
    }

    // 2) embeddings (also zeroes pad rows of x32/x16)
    embed_kernel<<<S_PAD, DIMC, 0, stream>>>(x, sp, te, x32, x16);

    const dim3 gM(S_PAD / 16, 1);
    const dim3 gF(S_PAD / 16, FFN_DIM / 256);
    const dim3 gA(S_PAD / 16, NHEADS);

    for (int l = 0; l < NLAYERS; ++l) {
        const size_t wo = (size_t)l * DIMC * DIMC;
        const size_t w1o = (size_t)l * DIMC * FFN_DIM;
        // QKV projections (f16 outputs; V transposed [H*DH][S_PAD])
        gemm_kernel<false, 1><<<gM, 128, 0, stream>>>(
            x16, WqT + wo, bq + l * DIMC, nullptr, q16, DIMC, DIMC);
        gemm_kernel<false, 1><<<gM, 128, 0, stream>>>(
            x16, WkT + wo, bk + l * DIMC, nullptr, k16, DIMC, DIMC);
        gemm_kernel<false, 2><<<gM, 128, 0, stream>>>(
            x16, WvT + wo, bv + l * DIMC, nullptr, vT16, DIMC, DIMC);
        // masked flash attention
        attn_kernel<<<gA, 32, 0, stream>>>(q16, k16, vT16, at16);
        // output projection (f32) + residual + LN1
        gemm_kernel<false, 0><<<gM, 128, 0, stream>>>(
            at16, WoT + wo, bo + l * DIMC, y32, nullptr, DIMC, DIMC);
        ln_residual_kernel<<<NS, DIMC, 0, stream>>>(
            x32, y32, ln1g + l * DIMC, ln1b + l * DIMC, x32, x16);
        // FFN
        gemm_kernel<true, 1><<<gF, 128, 0, stream>>>(
            x16, W1T + w1o, b1 + l * FFN_DIM, nullptr, h16, DIMC, FFN_DIM);
        gemm_kernel<false, 0><<<gM, 128, 0, stream>>>(
            h16, W2T + w1o, b2 + l * DIMC, y32, nullptr, FFN_DIM, DIMC);
        float* xo = (l == NLAYERS - 1) ? out : x32;
        ln_residual_kernel<<<NS, DIMC, 0, stream>>>(
            x32, y32, ln2g + l * DIMC, ln2b + l * DIMC, xo, x16);
    }
}
